// PointGCN_83726092468498
// MI455X (gfx1250) — matrix operand order repaired and checked by reading, not compiled, not run on hardware
//
#include <hip/hip_runtime.h>
#include <hip/hip_bf16.h>
#include <stdint.h>

typedef __attribute__((ext_vector_type(16))) __bf16          v16bf;
typedef __attribute__((ext_vector_type(16))) unsigned short  v16us;
typedef __attribute__((ext_vector_type(8)))  float           v8f;

#define WMMA_BF16(a, b, c) \
  __builtin_amdgcn_wmma_f32_16x16x32_bf16(false, (a), false, (b), (short)0, (c), false, false)

// CDNA5 async global->LDS copy (8B per lane), tracked by ASYNCcnt.
// VDST operand supplies the per-lane LDS byte offset; per ISA 10.2, the low 32
// bits of a generic LDS pointer are exactly that offset.
#define ASYNC_B64_TO_LDS(ldsoff, gaddr)                                         \
  asm volatile("global_load_async_to_lds_b64 %0, %1, off"                       \
               :: "v"(ldsoff), "v"(gaddr) : "memory")
#define WAIT_ASYNCCNT0() asm volatile("s_wait_asynccnt 0" ::: "memory")

__device__ __forceinline__ unsigned short f2bf(float f) {
  unsigned int u = __builtin_bit_cast(unsigned int, f);
  u += 0x7FFFu + ((u >> 16) & 1u);          // round-to-nearest-even
  return (unsigned short)(u >> 16);
}

// A-fragment (16x32 bf16): lane reads two contiguous 8-half (16B) runs of its row.
__device__ __forceinline__ v16bf load_a_lds(const unsigned short* base, int ldk, int lane) {
  int m  = lane & 15;
  int kb = (lane >> 4) * 8;
  const unsigned short* r = base + m * ldk + kb;
  v16us u;
#pragma unroll
  for (int j = 0; j < 8; ++j) { u[j] = r[j]; u[8 + j] = r[16 + j]; }
  return __builtin_bit_cast(v16bf, u);
}

// Same fragment loaded straight from a contiguous row-major bf16 matrix in global.
__device__ __forceinline__ v16bf load_a_glb(const unsigned short* __restrict__ mat,
                                            size_t row0, int ldk, int k0, int lane) {
  int m  = lane & 15;
  int kb = (lane >> 4) * 8;
  const unsigned short* r = mat + (row0 + m) * (size_t)ldk + k0 + kb;
  v16us u;
#pragma unroll
  for (int j = 0; j < 8; ++j) { u[j] = r[j]; u[8 + j] = r[16 + j]; }
  return __builtin_bit_cast(v16bf, u);
}

// B-fragment from pre-swizzled fragment-major weight buffer:
// wf[((kc*nnt + nt)*32 + lane)*16 + j] holds lane's 16 contiguous bf16 values.
__device__ __forceinline__ v16bf load_b_frag(const unsigned short* __restrict__ wf,
                                             int kc, int nt, int nnt, int lane) {
  const v16us* p = (const v16us*)wf;
  return __builtin_bit_cast(v16bf, p[(kc * nnt + nt) * 32 + lane]);
}

// ---------------- init: zero stats, zero segment-max keys, build bf16 mirror of x ----------------
__global__ void k_init(float* stats, unsigned int* aggr, const float* __restrict__ x,
                       unsigned short* __restrict__ x_bf, long total) {
  long i = (long)blockIdx.x * blockDim.x + threadIdx.x;
  if (i < 576) stats[i] = 0.f;
  if (i < total) { aggr[i] = 0u; x_bf[i] = f2bf(x[i]); }
}

// ---------------- convert weights to bf16, swizzled to WMMA-B fragment order ----------------
__device__ __forceinline__ void cvt_frag(const float* __restrict__ src,
                                         unsigned short* __restrict__ dst,
                                         int i, int Nn, int Ksrc) {
  int j    = i & 15;
  int lane = (i >> 4) & 31;
  int frag = i >> 9;
  int nnt  = Nn >> 4;
  int nt   = frag % nnt;
  int kc   = frag / nnt;
  int k    = kc * 32 + (lane >> 4) * 8 + (j & 7) + (j >> 3) * 16;
  int n    = nt * 16 + (lane & 15);
  float v  = (k < Ksrc) ? src[k * Nn + n] : 0.f;
  dst[i] = f2bf(v);
}

// f_w1 with input reordered to [x(128) | posdelta(3) | pad(29)]:
// padded row k <128 -> source row k+3 ; 128..130 -> k-128 ; else zero.
__device__ __forceinline__ void cvt_frag_fw1(const float* __restrict__ src,
                                             unsigned short* __restrict__ dst, int i) {
  int j    = i & 15;
  int lane = (i >> 4) & 31;
  int frag = i >> 9;
  int nt   = frag & 7;         // nnt = 8
  int kc   = frag >> 3;
  int k    = kc * 32 + (lane >> 4) * 8 + (j & 7) + (j >> 3) * 16;
  int n    = nt * 16 + (lane & 15);
  float v = 0.f;
  if (k < 128)      v = src[(k + 3) * 128 + n];
  else if (k < 131) v = src[(k - 128) * 128 + n];
  dst[i] = f2bf(v);
}

__global__ void k_cvt(const float* __restrict__ f_w1, const float* __restrict__ f_w2,
                      const float* __restrict__ g_w1, const float* __restrict__ g_w2,
                      const float* __restrict__ h_w1,
                      unsigned short* f_w1p, unsigned short* f_w2b,
                      unsigned short* g_w1b, unsigned short* g_w2b,
                      unsigned short* h_w1b) {
  int i = blockIdx.x * blockDim.x + threadIdx.x;
  if (i < 160 * 128) {
    cvt_frag_fw1(f_w1, f_w1p, i);                              // K=160 (permuted, padded)
  } else if (i < 160 * 128 + 16384) {
    cvt_frag(f_w2, f_w2b, i - 160 * 128, 128, 128);
  } else if (i < 160 * 128 + 2 * 16384) {
    cvt_frag(g_w1, g_w1b, i - 160 * 128 - 16384, 128, 128);
  } else if (i < 160 * 128 + 3 * 16384) {
    cvt_frag(g_w2, g_w2b, i - 160 * 128 - 2 * 16384, 128, 128);
  } else if (i < 160 * 128 + 3 * 16384 + 4096) {
    cvt_frag(h_w1, h_w1b, i - 160 * 128 - 3 * 16384, 32, 128); // K=128, N=32
  }
}

// ---------------- node hidden: t = x @ h_w1 + b1 (N x 32), + BN stats ----------------
__global__ __launch_bounds__(256) void k_h1(const unsigned short* __restrict__ x_bf,
    const unsigned short* __restrict__ wb, const float* __restrict__ bias,
    float* __restrict__ t_out, float* __restrict__ gsum, float* __restrict__ gsq,
    int n_tiles) {
  __shared__ float s_sum[32], s_sq[32];
  int wid = threadIdx.x >> 5, lane = threadIdx.x & 31;
  if (threadIdx.x < 32) { s_sum[threadIdx.x] = 0.f; s_sq[threadIdx.x] = 0.f; }
  __syncthreads();
  int tile = blockIdx.x * 8 + wid;
  if (tile < n_tiles) {
    int row0 = tile * 16;
    v8f c0 = {}, c1 = {};
#pragma unroll
    for (int kc = 0; kc < 4; ++kc) {
      v16bf a = load_a_glb(x_bf, (size_t)row0, 128, kc * 32, lane);
      c0 = WMMA_BF16(a, load_b_frag(wb, kc, 0, 2, lane), c0);
      c1 = WMMA_BF16(a, load_b_frag(wb, kc, 1, 2, lane), c1);
    }
    int ncol = lane & 15, mhi = (lane >> 4) * 8;
    float b0 = bias[ncol], b1v = bias[16 + ncol];
    float ls0 = 0.f, lq0 = 0.f, ls1 = 0.f, lq1 = 0.f;
#pragma unroll
    for (int v = 0; v < 8; ++v) {
      int row = row0 + mhi + v;
      float v0 = c0[v] + b0, v1 = c1[v] + b1v;
      t_out[(size_t)row * 32 + ncol]      = v0;
      t_out[(size_t)row * 32 + 16 + ncol] = v1;
      ls0 += v0; lq0 += v0 * v0; ls1 += v1; lq1 += v1 * v1;
    }
    atomicAdd(&s_sum[ncol], ls0);      atomicAdd(&s_sq[ncol], lq0);
    atomicAdd(&s_sum[16 + ncol], ls1); atomicAdd(&s_sq[16 + ncol], lq1);
  }
  __syncthreads();
  if (threadIdx.x < 32) {
    atomicAdd(&gsum[threadIdx.x], s_sum[threadIdx.x]);
    atomicAdd(&gsq[threadIdx.x], s_sq[threadIdx.x]);
  }
}

// ---------------- BN -> affine (scale, shift) ----------------
__global__ void k_bnscale(const float* __restrict__ sum, const float* __restrict__ sq,
                          const float* __restrict__ g, const float* __restrict__ be,
                          float* scale, float* shift, int F, float inv_cnt) {
  int i = threadIdx.x;
  if (i < F) {
    float m = sum[i] * inv_cnt;
    float v = fmaxf(sq[i] * inv_cnt - m * m, 0.f);
    float s = g[i] * rsqrtf(v + 1e-5f);
    scale[i] = s;
    shift[i] = be[i] - m * s;
  }
}

// ---------------- delta = relu(bn(t)) @ h_w2 + h_b2 (N x 3) ----------------
__global__ void k_delta(const float* __restrict__ t, const float* __restrict__ scale,
                        const float* __restrict__ shift, const float* __restrict__ w2,
                        const float* __restrict__ b2, float* __restrict__ delta, int n) {
  int i = blockIdx.x * blockDim.x + threadIdx.x;
  if (i >= n) return;
  float a0 = b2[0], a1 = b2[1], a2 = b2[2];
#pragma unroll
  for (int k = 0; k < 32; ++k) {
    float h = fmaxf(t[(size_t)i * 32 + k] * scale[k] + shift[k], 0.f);
    a0 += h * w2[k * 3 + 0];
    a1 += h * w2[k * 3 + 1];
    a2 += h * w2[k * 3 + 2];
  }
  delta[i * 3 + 0] = a0; delta[i * 3 + 1] = a1; delta[i * 3 + 2] = a2;
}

// build 16x160 bf16 edge-feature tile in LDS: [x_bf[src] (128) | posdelta (3) | 0 (29)]
// x rows copied with async global->LDS DMA (no VGPR transit); dst indices cached in LDS.
__device__ __forceinline__ void build_edge_tile(unsigned short* A, int e0, int lane,
    const unsigned short* __restrict__ x_bf, const float* __restrict__ pos,
    const float* __restrict__ delta, const int* __restrict__ edges, int* dstS) {
  unsigned ldsA = (unsigned)(uintptr_t)A;     // low 32 bits of generic LDS ptr = LDS offset
#pragma unroll 4
  for (int r = 0; r < 16; ++r) {
    int e = e0 + r;
    int s = edges[2 * e], d = edges[2 * e + 1];
    if (lane == 0) dstS[r] = d;
    unsigned long long g = (unsigned long long)(uintptr_t)(x_bf + (size_t)s * 128)
                         + (unsigned)(lane * 8);
    unsigned l = ldsA + (unsigned)(r * 320 + lane * 8);
    ASYNC_B64_TO_LDS(l, g);                   // 32 lanes x 8B = 256B row
    float v = 0.f;
    if (lane < 3) v = pos[s * 3 + lane] - pos[d * 3 + lane] + delta[d * 3 + lane];
    A[r * 160 + 128 + lane] = f2bf(v);
  }
  WAIT_ASYNCCNT0();                           // tile resident before fragment reads
}

// ---------------- edge pass 1: BN stats of e_in @ f_w1 + b1 ----------------
__global__ __launch_bounds__(256) void k_e1(const unsigned short* __restrict__ x_bf,
    const float* __restrict__ pos, const float* __restrict__ delta,
    const int* __restrict__ edges, const unsigned short* __restrict__ w1p,
    const float* __restrict__ b1, float* __restrict__ gsum, float* __restrict__ gsq,
    int n_tiles) {
  __shared__ unsigned short sA[8][16 * 160];
  __shared__ int sDst[8][16];
  __shared__ float s_sum[128], s_sq[128];
  int wid = threadIdx.x >> 5, lane = threadIdx.x & 31;
  if (threadIdx.x < 128) { s_sum[threadIdx.x] = 0.f; s_sq[threadIdx.x] = 0.f; }
  __syncthreads();
  int tile = blockIdx.x * 8 + wid;
  if (tile < n_tiles) {
    unsigned short* A = sA[wid];
    build_edge_tile(A, tile * 16, lane, x_bf, pos, delta, edges, sDst[wid]);
    v8f c[8] = {};
    for (int kc = 0; kc < 5; ++kc) {
      v16bf a = load_a_lds(A + kc * 32, 160, lane);
#pragma unroll
      for (int nt = 0; nt < 8; ++nt)
        c[nt] = WMMA_BF16(a, load_b_frag(w1p, kc, nt, 8, lane), c[nt]);
    }
    int ncol = lane & 15;
#pragma unroll
    for (int nt = 0; nt < 8; ++nt) {
      int col = nt * 16 + ncol;
      float bb = b1[col], ls = 0.f, lq = 0.f;
#pragma unroll
      for (int v = 0; v < 8; ++v) { float h = c[nt][v] + bb; ls += h; lq += h * h; }
      atomicAdd(&s_sum[col], ls);
      atomicAdd(&s_sq[col], lq);
    }
  }
  __syncthreads();
  if (threadIdx.x < 128) {
    atomicAdd(&gsum[threadIdx.x], s_sum[threadIdx.x]);
    atomicAdd(&gsq[threadIdx.x], s_sq[threadIdx.x]);
  }
}

// ---------------- edge pass 2: recompute, BN+relu, GEMM2, segment-max scatter ----------------
__global__ __launch_bounds__(256) void k_e2(const unsigned short* __restrict__ x_bf,
    const float* __restrict__ pos, const float* __restrict__ delta,
    const int* __restrict__ edges, const unsigned short* __restrict__ w1p,
    const float* __restrict__ b1, const float* __restrict__ scale,
    const float* __restrict__ shift, const unsigned short* __restrict__ w2b,
    const float* __restrict__ b2, unsigned int* __restrict__ aggr, int n_tiles) {
  __shared__ unsigned short sA[8][16 * 160];   // reused as 16x128 bf16 hidden tile
  __shared__ int sDst[8][16];
  int wid = threadIdx.x >> 5, lane = threadIdx.x & 31;
  int tile = blockIdx.x * 8 + wid;
  if (tile >= n_tiles) return;
  unsigned short* A = sA[wid];
  build_edge_tile(A, tile * 16, lane, x_bf, pos, delta, edges, sDst[wid]);
  v8f c[8] = {};
  for (int kc = 0; kc < 5; ++kc) {
    v16bf a = load_a_lds(A + kc * 32, 160, lane);
#pragma unroll
    for (int nt = 0; nt < 8; ++nt)
      c[nt] = WMMA_BF16(a, load_b_frag(w1p, kc, nt, 8, lane), c[nt]);
  }
  int ncol = lane & 15, mhi = (lane >> 4) * 8;
  // affine(BN)+relu, restage as bf16 A-tile (overwrites the dead e_in tile)
#pragma unroll
  for (int nt = 0; nt < 8; ++nt) {
    int col = nt * 16 + ncol;
    float sc = scale[col], sh = shift[col], bb = b1[col];
#pragma unroll
    for (int v = 0; v < 8; ++v) {
      float h = fmaxf((c[nt][v] + bb) * sc + sh, 0.f);
      A[(mhi + v) * 128 + col] = f2bf(h);
    }
  }
  v8f d[8] = {};
  for (int kc = 0; kc < 4; ++kc) {
    v16bf a = load_a_lds(A + kc * 32, 128, lane);
#pragma unroll
    for (int nt = 0; nt < 8; ++nt)
      d[nt] = WMMA_BF16(a, load_b_frag(w2b, kc, nt, 8, lane), d[nt]);
  }
#pragma unroll
  for (int v = 0; v < 8; ++v) {
    int dstn = sDst[wid][mhi + v];
    unsigned int* arow = aggr + (size_t)dstn * 128;
#pragma unroll
    for (int nt = 0; nt < 8; ++nt) {
      int col = nt * 16 + ncol;
      float msg = d[nt][v] + b2[col];
      unsigned int u = __builtin_bit_cast(unsigned int, msg);
      unsigned int key = (u & 0x80000000u) ? ~u : (u | 0x80000000u);  // order-preserving
      atomicMax(&arow[col], key);
    }
  }
}

// ---------------- decode segment-max keys -> bf16 aggr; non-finite / untouched -> 0 ----------------
__global__ void k_decode(const unsigned int* __restrict__ aggr,
                         unsigned short* __restrict__ aggr_bf, long total) {
  long i = (long)blockIdx.x * blockDim.x + threadIdx.x;
  if (i >= total) return;
  unsigned int key = aggr[i];
  unsigned int u = (key & 0x80000000u) ? (key & 0x7FFFFFFFu) : ~key;
  if ((u & 0x7F800000u) == 0x7F800000u) u = 0u;   // inf/nan (incl. untouched) -> 0
  aggr_bf[i] = f2bf(__builtin_bit_cast(float, u));
}

// ---------------- out pass 1: BN stats of aggr @ g_w1 + b1 ----------------
__global__ __launch_bounds__(256) void k_g1(const unsigned short* __restrict__ agg_bf,
    const unsigned short* __restrict__ w1b, const float* __restrict__ b1,
    float* __restrict__ gsum, float* __restrict__ gsq, int n_tiles) {
  __shared__ float s_sum[128], s_sq[128];
  int wid = threadIdx.x >> 5, lane = threadIdx.x & 31;
  if (threadIdx.x < 128) { s_sum[threadIdx.x] = 0.f; s_sq[threadIdx.x] = 0.f; }
  __syncthreads();
  int tile = blockIdx.x * 8 + wid;
  if (tile < n_tiles) {
    int row0 = tile * 16;
    v8f c[8] = {};
    for (int kc = 0; kc < 4; ++kc) {
      v16bf a = load_a_glb(agg_bf, (size_t)row0, 128, kc * 32, lane);
#pragma unroll
      for (int nt = 0; nt < 8; ++nt)
        c[nt] = WMMA_BF16(a, load_b_frag(w1b, kc, nt, 8, lane), c[nt]);
    }
    int ncol = lane & 15;
#pragma unroll
    for (int nt = 0; nt < 8; ++nt) {
      int col = nt * 16 + ncol;
      float bb = b1[col], ls = 0.f, lq = 0.f;
#pragma unroll
      for (int v = 0; v < 8; ++v) { float h = c[nt][v] + bb; ls += h; lq += h * h; }
      atomicAdd(&s_sum[col], ls);
      atomicAdd(&s_sq[col], lq);
    }
  }
  __syncthreads();
  if (threadIdx.x < 128) {
    atomicAdd(&gsum[threadIdx.x], s_sum[threadIdx.x]);
    atomicAdd(&gsq[threadIdx.x], s_sq[threadIdx.x]);
  }
}

// ---------------- out pass 2: recompute, BN+relu, GEMM2, out = x + . ----------------
__global__ __launch_bounds__(256) void k_g2(const unsigned short* __restrict__ agg_bf,
    const unsigned short* __restrict__ w1b, const float* __restrict__ b1,
    const float* __restrict__ scale, const float* __restrict__ shift,
    const unsigned short* __restrict__ w2b, const float* __restrict__ b2,
    const float* __restrict__ x, float* __restrict__ out, int n_tiles) {
  __shared__ unsigned short sH[8][16 * 128];
  int wid = threadIdx.x >> 5, lane = threadIdx.x & 31;
  int tile = blockIdx.x * 8 + wid;
  if (tile >= n_tiles) return;
  int row0 = tile * 16;
  v8f c[8] = {};
  for (int kc = 0; kc < 4; ++kc) {
    v16bf a = load_a_glb(agg_bf, (size_t)row0, 128, kc * 32, lane);
#pragma unroll
    for (int nt = 0; nt < 8; ++nt)
      c[nt] = WMMA_BF16(a, load_b_frag(w1b, kc, nt, 8, lane), c[nt]);
  }
  int ncol = lane & 15, mhi = (lane >> 4) * 8;
  unsigned short* H = sH[wid];
#pragma unroll
  for (int nt = 0; nt < 8; ++nt) {
    int col = nt * 16 + ncol;
    float sc = scale[col], sh = shift[col], bb = b1[col];
#pragma unroll
    for (int v = 0; v < 8; ++v) {
      float h = fmaxf((c[nt][v] + bb) * sc + sh, 0.f);
      H[(mhi + v) * 128 + col] = f2bf(h);
    }
  }
  v8f d[8] = {};
  for (int kc = 0; kc < 4; ++kc) {
    v16bf a = load_a_lds(H + kc * 32, 128, lane);
#pragma unroll
    for (int nt = 0; nt < 8; ++nt)
      d[nt] = WMMA_BF16(a, load_b_frag(w2b, kc, nt, 8, lane), d[nt]);
  }
#pragma unroll
  for (int v = 0; v < 8; ++v) {
    size_t row = (size_t)(row0 + mhi + v) * 128;
#pragma unroll
    for (int nt = 0; nt < 8; ++nt) {
      int col = nt * 16 + ncol;
      out[row + col] = x[row + col] + d[nt][v] + b2[col];
    }
  }
}

extern "C" void kernel_launch(void* const* d_in, const int* in_sizes, int n_in,
                              void* d_out, int out_size, void* d_ws, size_t ws_size,
                              hipStream_t stream) {
  const float* x    = (const float*)d_in[0];
  const float* pos  = (const float*)d_in[1];
  const int*   edges= (const int*)d_in[2];
  const float* h_w1 = (const float*)d_in[3];
  const float* h_b1 = (const float*)d_in[4];
  const float* h_g  = (const float*)d_in[5];
  const float* h_be = (const float*)d_in[6];
  const float* h_w2 = (const float*)d_in[7];
  const float* h_b2 = (const float*)d_in[8];
  const float* f_w1 = (const float*)d_in[9];
  const float* f_b1 = (const float*)d_in[10];
  const float* f_g  = (const float*)d_in[11];
  const float* f_be = (const float*)d_in[12];
  const float* f_w2 = (const float*)d_in[13];
  const float* f_b2 = (const float*)d_in[14];
  const float* g_w1 = (const float*)d_in[15];
  const float* g_b1 = (const float*)d_in[16];
  const float* g_g  = (const float*)d_in[17];
  const float* g_be = (const float*)d_in[18];
  const float* g_w2 = (const float*)d_in[19];
  const float* g_b2 = (const float*)d_in[20];
  float* out = (float*)d_out;

  const int N = in_sizes[0] / 128;
  const int E = in_sizes[2] / 2;

  uintptr_t p = (uintptr_t)d_ws;
  auto carve = [&](size_t bytes) -> void* {
    uintptr_t q = (p + 255) & ~(uintptr_t)255;
    p = q + bytes;
    return (void*)q;
  };
  float* stats  = (float*)carve(576 * 4);   // [h_sum32 h_sq32 f_sum128 f_sq128 g_sum128 g_sq128]
  float* affine = (float*)carve(576 * 4);   // same layout: scale/shift pairs
  unsigned short* f_w1p = (unsigned short*)carve(160 * 128 * 2);
  unsigned short* f_w2b = (unsigned short*)carve(128 * 128 * 2);
  unsigned short* g_w1b = (unsigned short*)carve(128 * 128 * 2);
  unsigned short* g_w2b = (unsigned short*)carve(128 * 128 * 2);
  unsigned short* h_w1b = (unsigned short*)carve(128 * 32 * 2);
  unsigned short* x_bf  = (unsigned short*)carve((size_t)N * 128 * 2);
  unsigned short* agg_bf= (unsigned short*)carve((size_t)N * 128 * 2);
  float*        t_h   = (float*)carve((size_t)N * 32 * 4);
  float*        delta = (float*)carve((size_t)N * 3 * 4);
  unsigned int* aggr  = (unsigned int*)carve((size_t)N * 128 * 4);
  (void)ws_size; (void)n_in; (void)out_size;

  const long tot = (long)N * 128;
  const int  ib  = (int)((tot + 255) / 256);
  const int  nt_n = N / 16, nb_n = (nt_n + 7) / 8;
  const int  nt_e = E / 16, nb_e = (nt_e + 7) / 8;

  k_init<<<ib, 256, 0, stream>>>(stats, aggr, x, x_bf, tot);
  k_cvt<<<288, 256, 0, stream>>>(f_w1, f_w2, g_w1, g_w2, h_w1,
                                 f_w1p, f_w2b, g_w1b, g_w2b, h_w1b);
  // node hidden MLP
  k_h1<<<nb_n, 256, 0, stream>>>(x_bf, h_w1b, h_b1, t_h, stats + 0, stats + 32, nt_n);
  k_bnscale<<<1, 128, 0, stream>>>(stats + 0, stats + 32, h_g, h_be,
                                   affine + 0, affine + 32, 32, 1.f / (float)N);
  k_delta<<<(N + 255) / 256, 256, 0, stream>>>(t_h, affine + 0, affine + 32,
                                               h_w2, h_b2, delta, N);
  // edge MLP + segment max
  k_e1<<<nb_e, 256, 0, stream>>>(x_bf, pos, delta, edges, f_w1p, f_b1,
                                 stats + 64, stats + 192, nt_e);
  k_bnscale<<<1, 128, 0, stream>>>(stats + 64, stats + 192, f_g, f_be,
                                   affine + 64, affine + 192, 128, 1.f / (float)E);
  k_e2<<<nb_e, 256, 0, stream>>>(x_bf, pos, delta, edges, f_w1p, f_b1,
                                 affine + 64, affine + 192, f_w2b, f_b2, aggr, nt_e);
  k_decode<<<ib, 256, 0, stream>>>(aggr, agg_bf, tot);
  // output MLP + residual
  k_g1<<<nb_n, 256, 0, stream>>>(agg_bf, g_w1b, g_b1,
                                 stats + 320, stats + 448, nt_n);
  k_bnscale<<<1, 128, 0, stream>>>(stats + 320, stats + 448, g_g, g_be,
                                   affine + 320, affine + 448, 128, 1.f / (float)N);
  k_g2<<<nb_n, 256, 0, stream>>>(agg_bf, g_w1b, g_b1,
                                 affine + 320, affine + 448, g_w2b, g_b2,
                                 x, out, nt_n);
}
